// GraphMamba_53652731462302
// MI455X (gfx1250) — matrix operand order repaired and checked by reading, not compiled, not run on hardware
//
#include <hip/hip_runtime.h>
#include <hip/hip_bf16.h>
#include <cstdint>

#define T_DIM 64
#define N_DIM 1024
#define P_DIM 256
#define H_DIM 256
#define E_DIM 512
#define S_DIM 16
#define R_DIM 16

typedef __attribute__((ext_vector_type(16))) __bf16 bf16x16;
typedef __attribute__((ext_vector_type(8)))  __bf16 bf16x8;
typedef __attribute__((ext_vector_type(8)))  float  f32x8;
typedef __attribute__((ext_vector_type(4)))  float  f32x4;

#define WMMA_BF16(acc, a, b) \
  (acc) = __builtin_amdgcn_wmma_f32_16x16x32_bf16(false, (a), false, (b), (short)0, (acc), false, false)

static __device__ __forceinline__ bf16x16 cat8(bf16x8 lo, bf16x8 hi) {
  return __builtin_shufflevector(lo, hi, 0,1,2,3,4,5,6,7,8,9,10,11,12,13,14,15);
}

static __device__ __forceinline__ bf16x16 loadA_bf16(const __bf16* __restrict__ A, int lda, int m0, int kc) {
  int lane = threadIdx.x & 31;
  int half = lane >> 4, mr = lane & 15;
  const __bf16* p = A + (size_t)(m0 + mr) * lda + kc + 8 * half;
  bf16x8 lo = *reinterpret_cast<const bf16x8*>(p);
  bf16x8 hi = *reinterpret_cast<const bf16x8*>(p + 16);
  return cat8(lo, hi);
}

static __device__ __forceinline__ bf16x16 loadB_bt(const __bf16* __restrict__ BT, int ldb, int n0, int kc) {
  int lane = threadIdx.x & 31;
  int hk = lane >> 4, nr = lane & 15;
  const __bf16* p = BT + (size_t)(n0 + nr) * ldb + kc + 8 * hk;
  bf16x8 lo = *reinterpret_cast<const bf16x8*>(p);
  bf16x8 hi = *reinterpret_cast<const bf16x8*>(p + 16);
  return cat8(lo, hi);
}

static __device__ __forceinline__ bf16x16 loadA_mask(const __bf16* __restrict__ G,
                                                     const float* __restrict__ Adj,
                                                     int ld, int m0, int kc) {
  int lane = threadIdx.x & 31;
  int half = lane >> 4, mr = lane & 15;
  size_t base = (size_t)(m0 + mr) * ld + kc + 8 * half;
  f32x4 a0 = *reinterpret_cast<const f32x4*>(Adj + base);
  f32x4 a1 = *reinterpret_cast<const f32x4*>(Adj + base + 4);
  f32x4 a2 = *reinterpret_cast<const f32x4*>(Adj + base + 16);
  f32x4 a3 = *reinterpret_cast<const f32x4*>(Adj + base + 20);
  bf16x16 r;
  if (G) {
    bf16x8 g0 = *reinterpret_cast<const bf16x8*>(G + base);
    bf16x8 g1 = *reinterpret_cast<const bf16x8*>(G + base + 16);
#pragma unroll
    for (int i = 0; i < 4; i++) {
      r[i]      = (__bf16)((float)g0[i]     * a0[i]);
      r[i + 4]  = (__bf16)((float)g0[i + 4] * a1[i]);
      r[i + 8]  = (__bf16)((float)g1[i]     * a2[i]);
      r[i + 12] = (__bf16)((float)g1[i + 4] * a3[i]);
    }
  } else {
#pragma unroll
    for (int i = 0; i < 4; i++) {
      r[i] = (__bf16)a0[i]; r[i + 4] = (__bf16)a1[i];
      r[i + 8] = (__bf16)a2[i]; r[i + 12] = (__bf16)a3[i];
    }
  }
  return r;
}

__global__ void k_cvt(const float* __restrict__ src, __bf16* __restrict__ dst, int n) {
  int i = blockIdx.x * blockDim.x + threadIdx.x;
  if (i < n) dst[i] = (__bf16)src[i];
}

__global__ void k_tcvt(const float* __restrict__ src, __bf16* __restrict__ dst, int R, int C) {
  int i = blockIdx.x * blockDim.x + threadIdx.x;
  if (i >= R * C) return;
  int r = i / C, c = i % C;
  dst[(size_t)c * R + r] = (__bf16)src[i];
}

__global__ void k_gemm_plain(const __bf16* __restrict__ A, int lda, size_t aStrideT,
                             const __bf16* __restrict__ BT, int ldb,
                             const float* __restrict__ bias,
                             __bf16* outN, int ldoN, size_t oStrideT,
                             __bf16* outT, int ldoT, size_t oTStrideT,
                             int M, int Nn, int K) {
  int w = (int)((blockIdx.x * blockDim.x + threadIdx.x) >> 5);
  int ntiles = Nn >> 4;
  int tiles = (M >> 4) * ntiles;
  if (w >= tiles) return;
  int m0 = (w / ntiles) << 4, n0 = (w % ntiles) << 4;
  const __bf16* At = A + (size_t)blockIdx.z * aStrideT;
  f32x8 acc = {0,0,0,0,0,0,0,0};
  for (int kc = 0; kc < K; kc += 32) {
    bf16x16 a = loadA_bf16(At, lda, m0, kc);
    bf16x16 b = loadB_bt(BT, ldb, n0, kc);
    WMMA_BF16(acc, a, b);
  }
  int lane = threadIdx.x & 31, half = lane >> 4, nr = lane & 15;
  float bv = bias ? bias[n0 + nr] : 0.f;
  if (outN) {
    __bf16* o = outN + (size_t)blockIdx.z * oStrideT;
#pragma unroll
    for (int r = 0; r < 8; r++) o[(size_t)(m0 + r + 8 * half) * ldoN + n0 + nr] = (__bf16)(acc[r] + bv);
  }
  if (outT) {
    __bf16* o = outT + (size_t)blockIdx.z * oTStrideT;
#pragma unroll
    for (int r = 0; r < 8; r++) o[(size_t)(n0 + nr) * ldoT + m0 + r + 8 * half] = (__bf16)(acc[r] + bv);
  }
}

__global__ void k_gemm_gates(const __bf16* __restrict__ A, int lda,
                             const __bf16* __restrict__ BT, int ldb,
                             const float* __restrict__ gbp, float invscale,
                             __bf16* __restrict__ out, int ldo,
                             int M, int Nn, int K) {
  int w = (int)((blockIdx.x * blockDim.x + threadIdx.x) >> 5);
  int ntiles = Nn >> 4;
  int tiles = (M >> 4) * ntiles;
  if (w >= tiles) return;
  int m0 = (w / ntiles) << 4, n0 = (w % ntiles) << 4;
  f32x8 acc = {0,0,0,0,0,0,0,0};
  for (int kc = 0; kc < K; kc += 32) {
    bf16x16 a = loadA_bf16(A, lda, m0, kc);
    bf16x16 b = loadB_bt(BT, ldb, n0, kc);
    WMMA_BF16(acc, a, b);
  }
  float gb = gbp[0];
  int lane = threadIdx.x & 31, half = lane >> 4, nr = lane & 15;
#pragma unroll
  for (int r = 0; r < 8; r++) {
    float v = acc[r] * invscale + gb;
    out[(size_t)(m0 + r + 8 * half) * ldo + n0 + nr] = (__bf16)(1.f / (1.f + __expf(-v)));
  }
}

__global__ void k_gemm_maskA(const float* __restrict__ adjSeq, size_t adjStrideT,
                             const __bf16* __restrict__ gates,
                             const __bf16* __restrict__ BT, int ldb, size_t bStrideT,
                             __bf16* __restrict__ out, int ldo, size_t oStrideT,
                             int M, int Nn, int K) {
  int w = (int)((blockIdx.x * blockDim.x + threadIdx.x) >> 5);
  int ntiles = Nn >> 4;
  int tiles = (M >> 4) * ntiles;
  if (w >= tiles) return;
  int m0 = (w / ntiles) << 4, n0 = (w % ntiles) << 4;
  const float* adjT = adjSeq + (size_t)blockIdx.z * adjStrideT;
  const __bf16* BTt = BT + (size_t)blockIdx.z * bStrideT;
  f32x8 acc = {0,0,0,0,0,0,0,0};
  for (int kc = 0; kc < K; kc += 32) {
    bf16x16 a = loadA_mask(gates, adjT, K, m0, kc);
    bf16x16 b = loadB_bt(BTt, ldb, n0, kc);
    WMMA_BF16(acc, a, b);
  }
  __bf16* o = out + (size_t)blockIdx.z * oStrideT;
  int lane = threadIdx.x & 31, half = lane >> 4, nr = lane & 15;
#pragma unroll
  for (int r = 0; r < 8; r++) o[(size_t)(m0 + r + 8 * half) * ldo + n0 + nr] = (__bf16)acc[r];
}

__global__ void k_gemm_dual_relu(const __bf16* __restrict__ A1, int lda1, size_t a1StrideT,
                                 const __bf16* __restrict__ A2, int lda2, size_t a2StrideT,
                                 int K1,
                                 const __bf16* __restrict__ BT, int ldb,
                                 const float* __restrict__ bias,
                                 float* __restrict__ out, int ldo, size_t oStrideT,
                                 int M, int Nn, int K) {
  int w = (int)((blockIdx.x * blockDim.x + threadIdx.x) >> 5);
  int ntiles = Nn >> 4;
  int tiles = (M >> 4) * ntiles;
  if (w >= tiles) return;
  int m0 = (w / ntiles) << 4, n0 = (w % ntiles) << 4;
  const __bf16* A1t = A1 + (size_t)blockIdx.z * a1StrideT;
  const __bf16* A2t = A2 + (size_t)blockIdx.z * a2StrideT;
  f32x8 acc = {0,0,0,0,0,0,0,0};
  for (int kc = 0; kc < K; kc += 32) {
    bf16x16 a = (kc < K1) ? loadA_bf16(A1t, lda1, m0, kc) : loadA_bf16(A2t, lda2, m0, kc - K1);
    bf16x16 b = loadB_bt(BT, ldb, n0, kc);
    WMMA_BF16(acc, a, b);
  }
  float* o = out + (size_t)blockIdx.z * oStrideT;
  int lane = threadIdx.x & 31, half = lane >> 4, nr = lane & 15;
  float bv = bias[n0 + nr];
#pragma unroll
  for (int r = 0; r < 8; r++) o[(size_t)(m0 + r + 8 * half) * ldo + n0 + nr] = fmaxf(acc[r] + bv, 0.f);
}

__global__ void k_gemm_f32out(const __bf16* __restrict__ A, int lda,
                              const __bf16* __restrict__ BT, int ldb,
                              const float* __restrict__ bias,
                              float* __restrict__ out, int ldo,
                              int M, int Nn, int K) {
  int w = (int)((blockIdx.x * blockDim.x + threadIdx.x) >> 5);
  int ntiles = Nn >> 4;
  int tiles = (M >> 4) * ntiles;
  if (w >= tiles) return;
  int m0 = (w / ntiles) << 4, n0 = (w % ntiles) << 4;
  f32x8 acc = {0,0,0,0,0,0,0,0};
  for (int kc = 0; kc < K; kc += 32) {
    bf16x16 a = loadA_bf16(A, lda, m0, kc);
    bf16x16 b = loadB_bt(BT, ldb, n0, kc);
    WMMA_BF16(acc, a, b);
  }
  int lane = threadIdx.x & 31, half = lane >> 4, nr = lane & 15;
  float bv = bias[n0 + nr];
#pragma unroll
  for (int r = 0; r < 8; r++) out[(size_t)(m0 + r + 8 * half) * ldo + n0 + nr] = acc[r] + bv;
}

__global__ void k_gemm_win(const __bf16* __restrict__ A,
                           const __bf16* __restrict__ BT,
                           const float* __restrict__ bias,
                           float* __restrict__ uF, __bf16* __restrict__ uB,
                           float* __restrict__ sgF,
                           int M, int Nn, int K) {
  int w = (int)((blockIdx.x * blockDim.x + threadIdx.x) >> 5);
  int ntiles = Nn >> 4;
  int tiles = (M >> 4) * ntiles;
  if (w >= tiles) return;
  int m0 = (w / ntiles) << 4, n0 = (w % ntiles) << 4;
  f32x8 acc = {0,0,0,0,0,0,0,0};
  for (int kc = 0; kc < K; kc += 32) {
    bf16x16 a = loadA_bf16(A, H_DIM, m0, kc);
    bf16x16 b = loadB_bt(BT, H_DIM, n0, kc);
    WMMA_BF16(acc, a, b);
  }
  int lane = threadIdx.x & 31, half = lane >> 4, nr = lane & 15;
  int col = n0 + nr;
  float bv = bias[col];
  if (n0 < E_DIM) {
#pragma unroll
    for (int r = 0; r < 8; r++) {
      float v = acc[r] + bv;
      float s = v / (1.f + __expf(-v));
      size_t idx = (size_t)(m0 + r + 8 * half) * E_DIM + col;
      uF[idx] = s; uB[idx] = (__bf16)s;
    }
  } else {
#pragma unroll
    for (int r = 0; r < 8; r++) {
      float v = acc[r] + bv;
      sgF[(size_t)(m0 + r + 8 * half) * E_DIM + (col - E_DIM)] = v / (1.f + __expf(-v));
    }
  }
}

__global__ void k_gemm_out(const __bf16* __restrict__ A,
                           const __bf16* __restrict__ BT,
                           const float* __restrict__ bias,
                           const float* __restrict__ res,
                           float* __restrict__ dout, int M) {
  int w = (int)((blockIdx.x * blockDim.x + threadIdx.x) >> 5);
  const int ntiles = H_DIM >> 4;
  int tiles = (M >> 4) * ntiles;
  if (w >= tiles) return;
  int m0 = (w / ntiles) << 4, n0 = (w % ntiles) << 4;
  f32x8 acc = {0,0,0,0,0,0,0,0};
  for (int kc = 0; kc < E_DIM; kc += 32) {
    bf16x16 a = loadA_bf16(A, E_DIM, m0, kc);
    bf16x16 b = loadB_bt(BT, E_DIM, n0, kc);
    WMMA_BF16(acc, a, b);
  }
  int lane = threadIdx.x & 31, half = lane >> 4, nr = lane & 15;
  int h = n0 + nr;
  float bv = bias[h];
#pragma unroll
  for (int r = 0; r < 8; r++) {
    int row = m0 + r + 8 * half;       // row = n*T + t
    int nn = row >> 6, tt = row & 63;  // T = 64
    float v = acc[r] + bv + res[(size_t)row * H_DIM + h];
    dout[((size_t)tt * N_DIM + nn) * H_DIM + h] = v;
  }
}

__global__ void k_ln(const float* __restrict__ in, int rows,
                     const float* __restrict__ gamma, const float* __restrict__ beta,
                     float* __restrict__ outF, __bf16* __restrict__ outB, int remap) {
  int w = (int)((blockIdx.x * blockDim.x + threadIdx.x) >> 5);
  if (w >= rows) return;
  int lane = threadIdx.x & 31;
  const float* p = in + (size_t)w * H_DIM + lane * 8;
  f32x4 x0 = *reinterpret_cast<const f32x4*>(p);
  f32x4 x1 = *reinterpret_cast<const f32x4*>(p + 4);
  float s = 0.f, q = 0.f;
#pragma unroll
  for (int i = 0; i < 4; i++) { s += x0[i] + x1[i]; q += x0[i]*x0[i] + x1[i]*x1[i]; }
  for (int m = 1; m < 32; m <<= 1) { s += __shfl_xor(s, m); q += __shfl_xor(q, m); }
  float mu = s * (1.f / H_DIM);
  float var = q * (1.f / H_DIM) - mu * mu;
  float inv = rsqrtf(var + 1e-5f);
  size_t orow = (size_t)w;
  if (remap) { int t = w / N_DIM, n = w % N_DIM; orow = (size_t)n * T_DIM + t; }
  f32x4 ga = *reinterpret_cast<const f32x4*>(gamma + lane * 8);
  f32x4 gh = *reinterpret_cast<const f32x4*>(gamma + lane * 8 + 4);
  f32x4 ba = *reinterpret_cast<const f32x4*>(beta + lane * 8);
  f32x4 bh = *reinterpret_cast<const f32x4*>(beta + lane * 8 + 4);
  f32x4 y0, y1;
#pragma unroll
  for (int i = 0; i < 4; i++) {
    y0[i] = (x0[i] - mu) * inv * ga[i] + ba[i];
    y1[i] = (x1[i] - mu) * inv * gh[i] + bh[i];
  }
  if (outF) {
    *reinterpret_cast<f32x4*>(outF + orow * H_DIM + lane * 8) = y0;
    *reinterpret_cast<f32x4*>(outF + orow * H_DIM + lane * 8 + 4) = y1;
  }
  if (outB) {
    bf16x8 ob;
#pragma unroll
    for (int i = 0; i < 4; i++) { ob[i] = (__bf16)y0[i]; ob[i + 4] = (__bf16)y1[i]; }
    *reinterpret_cast<bf16x8*>(outB + orow * H_DIM + lane * 8) = ob;
  }
}

__global__ void k_delta(const float* __restrict__ d1, const float* __restrict__ Wdt,
                        const float* __restrict__ bdt, float* __restrict__ delta, size_t n) {
  size_t i = (size_t)blockIdx.x * blockDim.x + threadIdx.x;
  if (i >= n) return;
  size_t row = i >> 9; int e = (int)(i & (E_DIM - 1));
  const float* dp = d1 + row * R_DIM;
  float acc = bdt[e];
#pragma unroll
  for (int r = 0; r < R_DIM; r++) acc += dp[r] * Wdt[r * E_DIM + e];
  delta[i] = (acc > 20.f) ? acc : log1pf(__expf(acc));
}

__global__ void k_scan(const float* __restrict__ u, const float* __restrict__ delta,
                       const float* __restrict__ Bm, const float* __restrict__ Cm,
                       const float* __restrict__ Alog, float* __restrict__ y) {
  int n = blockIdx.x;
  int e = blockIdx.y * blockDim.x + threadIdx.x;
  float a[S_DIM], h[S_DIM];
#pragma unroll
  for (int s = 0; s < S_DIM; s++) { a[s] = -__expf(Alog[e * S_DIM + s]); h[s] = 0.f; }
  for (int t = 0; t < T_DIM; t++) {
    size_t row = (size_t)n * T_DIM + t;
    float dt = delta[row * E_DIM + e];
    float ut = u[row * E_DIM + e];
    const f32x4* Bp = reinterpret_cast<const f32x4*>(Bm + row * S_DIM);
    const f32x4* Cp = reinterpret_cast<const f32x4*>(Cm + row * S_DIM);
    float acc = 0.f;
#pragma unroll
    for (int s4 = 0; s4 < 4; s4++) {
      f32x4 bv = Bp[s4], cv = Cp[s4];
#pragma unroll
      for (int j = 0; j < 4; j++) {
        int s = s4 * 4 + j;
        h[s] = __expf(dt * a[s]) * h[s] + dt * bv[j] * ut;
        acc += cv[j] * h[s];
      }
    }
    y[row * E_DIM + e] = acc;
  }
}

__global__ void k_yc(const float* __restrict__ y, const float* __restrict__ u,
                     const float* __restrict__ sg, const float* __restrict__ D,
                     __bf16* __restrict__ yc, size_t n) {
  size_t i = (size_t)blockIdx.x * blockDim.x + threadIdx.x;
  if (i >= n) return;
  int e = (int)(i & (E_DIM - 1));
  yc[i] = (__bf16)((y[i] + u[i] * D[e]) * sg[i]);
}

extern "C" void kernel_launch(void* const* d_in, const int* in_sizes, int n_in,
                              void* d_out, int out_size, void* d_ws, size_t ws_size,
                              hipStream_t stream) {
  (void)in_sizes; (void)n_in; (void)out_size; (void)ws_size;
  const float* adj     = (const float*)d_in[0];
  const float* pos     = (const float*)d_in[1];
  const float* g1Wm    = (const float*)d_in[2];
  const float* g1bm    = (const float*)d_in[3];
  const float* g1Wq    = (const float*)d_in[4];
  const float* g1Wk    = (const float*)d_in[5];
  const float* g1gb    = (const float*)d_in[6];
  const float* g1Wu    = (const float*)d_in[7];
  const float* g1bu    = (const float*)d_in[8];
  const float* g1lg    = (const float*)d_in[9];
  const float* g1lb    = (const float*)d_in[10];
  const float* g2Wm    = (const float*)d_in[11];
  const float* g2bm    = (const float*)d_in[12];
  const float* g2Wu    = (const float*)d_in[13];
  const float* g2bu    = (const float*)d_in[14];
  const float* g2lg    = (const float*)d_in[15];
  const float* g2lb    = (const float*)d_in[16];
  const float* mlg     = (const float*)d_in[17];
  const float* mlb     = (const float*)d_in[18];
  const float* mWin    = (const float*)d_in[19];
  const float* mbin    = (const float*)d_in[20];
  const float* mWdelta = (const float*)d_in[21];
  const float* mbdelta = (const float*)d_in[22];
  const float* mWdt    = (const float*)d_in[23];
  const float* mbdt    = (const float*)d_in[24];
  const float* mWB     = (const float*)d_in[25];
  const float* mbB     = (const float*)d_in[26];
  const float* mWC     = (const float*)d_in[27];
  const float* mbC     = (const float*)d_in[28];
  const float* mAlog   = (const float*)d_in[29];
  const float* mD      = (const float*)d_in[30];
  const float* mWout   = (const float*)d_in[31];
  const float* mbout   = (const float*)d_in[32];
  float* dout = (float*)d_out;

  const int NT = N_DIM * T_DIM;
  char* base = (char*)d_ws;
  size_t off = 0;
  auto alloc = [&](size_t bytes) -> char* {
    char* p = base + off; off += (bytes + 255) & ~(size_t)255; return p;
  };

  __bf16* posb   = (__bf16*)alloc((size_t)N_DIM * P_DIM * 2);
  __bf16* WmT    = (__bf16*)alloc((size_t)H_DIM * P_DIM * 2);
  __bf16* WqT    = (__bf16*)alloc((size_t)H_DIM * H_DIM * 2);
  __bf16* WkT    = (__bf16*)alloc((size_t)H_DIM * H_DIM * 2);
  __bf16* Wu1T   = (__bf16*)alloc((size_t)H_DIM * (P_DIM + H_DIM) * 2);
  __bf16* Wm2T   = (__bf16*)alloc((size_t)H_DIM * H_DIM * 2);
  __bf16* Wu2T   = (__bf16*)alloc((size_t)H_DIM * (2 * H_DIM) * 2);
  __bf16* WinT   = (__bf16*)alloc((size_t)(2 * E_DIM) * H_DIM * 2);
  __bf16* WdT    = (__bf16*)alloc((size_t)R_DIM * E_DIM * 2);
  __bf16* WBT    = (__bf16*)alloc((size_t)S_DIM * E_DIM * 2);
  __bf16* WCT    = (__bf16*)alloc((size_t)S_DIM * E_DIM * 2);
  __bf16* WoutT  = (__bf16*)alloc((size_t)H_DIM * E_DIM * 2);
  __bf16* msgsb  = (__bf16*)alloc((size_t)N_DIM * H_DIM * 2);
  __bf16* msgsTb = (__bf16*)alloc((size_t)H_DIM * N_DIM * 2);
  __bf16* qb     = (__bf16*)alloc((size_t)N_DIM * H_DIM * 2);
  __bf16* kb     = (__bf16*)alloc((size_t)N_DIM * H_DIM * 2);
  __bf16* gatesb = (__bf16*)alloc((size_t)N_DIM * N_DIM * 2);
  __bf16* aggb   = (__bf16*)alloc((size_t)T_DIM * N_DIM * H_DIM * 2);
  float*  xpre   = (float*) alloc((size_t)T_DIM * N_DIM * H_DIM * 4);
  __bf16* x1b    = (__bf16*)alloc((size_t)T_DIM * N_DIM * H_DIM * 2);
  __bf16* m2T    = (__bf16*)alloc((size_t)T_DIM * H_DIM * N_DIM * 2);
  float*  xf     = (float*) alloc((size_t)NT * H_DIM * 4);
  __bf16* xnb    = (__bf16*)alloc((size_t)NT * H_DIM * 2);
  float*  uF     = (float*) alloc((size_t)NT * E_DIM * 4);
  float*  sgF    = (float*) alloc((size_t)NT * E_DIM * 4);
  __bf16* uB     = (__bf16*)alloc((size_t)NT * E_DIM * 2);
  float*  d1     = (float*) alloc((size_t)NT * R_DIM * 4);
  float*  deltaF = (float*) alloc((size_t)NT * E_DIM * 4);
  float*  Bmb    = (float*) alloc((size_t)NT * S_DIM * 4);
  float*  Cmb    = (float*) alloc((size_t)NT * S_DIM * 4);
  float*  yF     = (float*) alloc((size_t)NT * E_DIM * 4);
  __bf16* ycb    = (__bf16*)alloc((size_t)NT * E_DIM * 2);

  const int THR = 256;
  auto gemmBlocks = [](int M, int Nn) { return (((M / 16) * (Nn / 16)) + 7) / 8; };
  auto tcvt = [&](const float* s, __bf16* d, int R, int C) {
    k_tcvt<<<dim3((R * C + THR - 1) / THR), THR, 0, stream>>>(s, d, R, C);
  };

  // conversions (once)
  k_cvt<<<dim3((N_DIM * P_DIM + THR - 1) / THR), THR, 0, stream>>>(pos, posb, N_DIM * P_DIM);
  tcvt(g1Wm, WmT, P_DIM, H_DIM);
  tcvt(g1Wq, WqT, H_DIM, H_DIM);
  tcvt(g1Wk, WkT, H_DIM, H_DIM);
  tcvt(g1Wu, Wu1T, P_DIM + H_DIM, H_DIM);
  tcvt(g2Wm, Wm2T, H_DIM, H_DIM);
  tcvt(g2Wu, Wu2T, 2 * H_DIM, H_DIM);
  tcvt(mWin, WinT, H_DIM, 2 * E_DIM);
  tcvt(mWdelta, WdT, E_DIM, R_DIM);
  tcvt(mWB, WBT, E_DIM, S_DIM);
  tcvt(mWC, WCT, E_DIM, S_DIM);
  tcvt(mWout, WoutT, E_DIM, H_DIM);

  // T-invariant attention pieces
  k_gemm_plain<<<dim3(gemmBlocks(N_DIM, H_DIM)), THR, 0, stream>>>(
      posb, P_DIM, 0, WmT, P_DIM, g1bm,
      msgsb, H_DIM, 0, msgsTb, N_DIM, 0, N_DIM, H_DIM, P_DIM);
  k_gemm_plain<<<dim3(gemmBlocks(N_DIM, H_DIM)), THR, 0, stream>>>(
      msgsb, H_DIM, 0, WqT, H_DIM, nullptr,
      qb, H_DIM, 0, nullptr, 0, 0, N_DIM, H_DIM, H_DIM);
  k_gemm_plain<<<dim3(gemmBlocks(N_DIM, H_DIM)), THR, 0, stream>>>(
      msgsb, H_DIM, 0, WkT, H_DIM, nullptr,
      kb, H_DIM, 0, nullptr, 0, 0, N_DIM, H_DIM, H_DIM);
  k_gemm_gates<<<dim3(gemmBlocks(N_DIM, N_DIM)), THR, 0, stream>>>(
      qb, H_DIM, kb, H_DIM, g1gb, 1.0f / 16.0f, gatesb, N_DIM, N_DIM, N_DIM, H_DIM);

  // graph encoder, batched over T
  dim3 gNH(gemmBlocks(N_DIM, H_DIM), 1, T_DIM);
  k_gemm_maskA<<<gNH, THR, 0, stream>>>(
      adj, (size_t)N_DIM * N_DIM, gatesb, msgsTb, N_DIM, 0,
      aggb, H_DIM, (size_t)N_DIM * H_DIM, N_DIM, H_DIM, N_DIM);
  k_gemm_dual_relu<<<gNH, THR, 0, stream>>>(
      posb, P_DIM, 0, aggb, H_DIM, (size_t)N_DIM * H_DIM, P_DIM,
      Wu1T, P_DIM + H_DIM, g1bu,
      xpre, H_DIM, (size_t)N_DIM * H_DIM, N_DIM, H_DIM, P_DIM + H_DIM);
  k_ln<<<dim3(NT / 8), THR, 0, stream>>>(xpre, NT, g1lg, g1lb, nullptr, x1b, 0);
  k_gemm_plain<<<gNH, THR, 0, stream>>>(
      x1b, H_DIM, (size_t)N_DIM * H_DIM, Wm2T, H_DIM, g2bm,
      nullptr, 0, 0, m2T, N_DIM, (size_t)H_DIM * N_DIM, N_DIM, H_DIM, H_DIM);
  k_gemm_maskA<<<gNH, THR, 0, stream>>>(
      adj, (size_t)N_DIM * N_DIM, nullptr, m2T, N_DIM, (size_t)H_DIM * N_DIM,
      aggb, H_DIM, (size_t)N_DIM * H_DIM, N_DIM, H_DIM, N_DIM);
  k_gemm_dual_relu<<<gNH, THR, 0, stream>>>(
      x1b, H_DIM, (size_t)N_DIM * H_DIM, aggb, H_DIM, (size_t)N_DIM * H_DIM, H_DIM,
      Wu2T, 2 * H_DIM, g2bu,
      xpre, H_DIM, (size_t)N_DIM * H_DIM, N_DIM, H_DIM, 2 * H_DIM);
  k_ln<<<dim3(NT / 8), THR, 0, stream>>>(xpre, NT, g2lg, g2lb, xf, nullptr, 1);

  // Mamba block
  k_ln<<<dim3(NT / 8), THR, 0, stream>>>(xf, NT, mlg, mlb, nullptr, xnb, 0);
  k_gemm_win<<<dim3(gemmBlocks(NT, 2 * E_DIM)), THR, 0, stream>>>(
      xnb, WinT, mbin, uF, uB, sgF, NT, 2 * E_DIM, H_DIM);
  k_gemm_f32out<<<dim3(gemmBlocks(NT, R_DIM)), THR, 0, stream>>>(
      uB, E_DIM, WdT, E_DIM, mbdelta, d1, R_DIM, NT, R_DIM, E_DIM);
  {
    size_t n = (size_t)NT * E_DIM;
    k_delta<<<dim3((unsigned)((n + THR - 1) / THR)), THR, 0, stream>>>(d1, mWdt, mbdt, deltaF, n);
  }
  k_gemm_f32out<<<dim3(gemmBlocks(NT, S_DIM)), THR, 0, stream>>>(
      uB, E_DIM, WBT, E_DIM, mbB, Bmb, S_DIM, NT, S_DIM, E_DIM);
  k_gemm_f32out<<<dim3(gemmBlocks(NT, S_DIM)), THR, 0, stream>>>(
      uB, E_DIM, WCT, E_DIM, mbC, Cmb, S_DIM, NT, S_DIM, E_DIM);
  k_scan<<<dim3(N_DIM, E_DIM / THR), THR, 0, stream>>>(uF, deltaF, Bmb, Cmb, mAlog, yF);
  {
    size_t n = (size_t)NT * E_DIM;
    k_yc<<<dim3((unsigned)((n + THR - 1) / THR)), THR, 0, stream>>>(yF, uF, sgF, mD, ycb, n);
  }
  k_gemm_out<<<dim3(gemmBlocks(NT, H_DIM)), THR, 0, stream>>>(
      ycb, WoutT, mbout, xf, dout, NT);
}
